// MAGAC_9474697855320
// MI455X (gfx1250) — matrix operand order repaired and checked by reading, not compiled, not run on hardware
//
#include <hip/hip_runtime.h>

// ---- problem constants (match reference) ----
#define B_   32
#define N_   1024
#define L_   128
#define DE   16
#define H_   4
#define C_   (B_ * L_)        // 4096 columns = (b, l)

typedef _Float16 f16;
typedef _Float16 half8 __attribute__((ext_vector_type(8)));
typedef _Float16 v16h  __attribute__((ext_vector_type(16)));
typedef float    v8f   __attribute__((ext_vector_type(8)));

// =====================================================================
// CDNA5 async global->LDS copy (16B per lane), tracked by ASYNCcnt.
// lds  = LDS byte address (low 32 bits of generic pointer)
// gptr = global address
// =====================================================================
__device__ __forceinline__ void async_copy_b128(unsigned lds, const f16* gptr) {
    asm volatile("global_load_async_to_lds_b128 %0, %1, off"
                 :: "v"(lds), "v"(gptr) : "memory");
}

// =====================================================================
// Kernel 1: Q/K projections + bias filter.  grid=N_, block=128
// =====================================================================
__global__ void prep_qk_kernel(const float* __restrict__ psi_emb,
                               const float* __restrict__ Wq,
                               const float* __restrict__ Wk,
                               const float* __restrict__ fb,
                               float* __restrict__ Q, float* __restrict__ Kt,
                               float* __restrict__ bf) {
    int n = blockIdx.x, t = threadIdx.x;
    __shared__ float p[DE];
    if (t < DE) p[t] = psi_emb[n * DE + t];
    __syncthreads();
    if (t < 64) {
        int h = t >> 4, d = t & 15;
        float aq = 0.f, ak = 0.f;
#pragma unroll
        for (int d0 = 0; d0 < DE; ++d0) {
            float pv = p[d0];
            aq += pv * Wq[(d0 * H_ + h) * DE + d];
            ak += pv * Wk[(d0 * H_ + h) * DE + d];
        }
        Q [(n * H_ + h) * DE + d] = aq;
        Kt[(n * H_ + h) * DE + d] = ak;
    } else if (t < 64 + H_) {
        int h = t - 64;
        float a = 0.f;
#pragma unroll
        for (int d = 0; d < DE; ++d) a += p[d] * fb[h * DE + d];
        bf[h * N_ + n] = a;
    }
}

// =====================================================================
// Kernel 2: A_base row softmax.  grid=N_, block=256 (4 cols / thread)
// =====================================================================
__global__ void abase_kernel(const float* __restrict__ psi_emb,
                             const float* __restrict__ psi_s,
                             float* __restrict__ Abase) {
    int n = blockIdx.x, t = threadIdx.x;
    __shared__ float pn[DE];
    __shared__ float red[256];
    if (t < DE) pn[t] = psi_emb[n * DE + t];
    __syncthreads();
    float psi = *psi_s;
    float ev[4], lsum = 0.f;
#pragma unroll
    for (int i = 0; i < 4; ++i) {
        int m = t + i * 256;
        float d2 = 0.f;
#pragma unroll
        for (int d = 0; d < DE; ++d) {
            float df = pn[d] - psi_emb[m * DE + d];
            d2 += df * df;
        }
        float e = __expf(__expf(-psi * d2));   // softmax numerator of exp(-psi*d2)
        ev[i] = e; lsum += e;
    }
    red[t] = lsum; __syncthreads();
    for (int s = 128; s > 0; s >>= 1) {
        if (t < s) red[t] += red[t + s];
        __syncthreads();
    }
    float inv = 1.f / red[0];
#pragma unroll
    for (int i = 0; i < 4; ++i) Abase[(size_t)n * N_ + t + i * 256] = ev[i] * inv;
}

// =====================================================================
// Kernel 3: per-head attention softmax + sigmoid blend -> A_h (f16)
// =====================================================================
__global__ void ah_kernel(const float* __restrict__ Q,
                          const float* __restrict__ Kt,
                          const float* __restrict__ Abase,
                          const float* __restrict__ alpha_s,
                          f16* __restrict__ Ah, int h) {
    int n = blockIdx.x, t = threadIdx.x;
    __shared__ float qn[DE];
    __shared__ float red[256];
    if (t < DE) qn[t] = Q[(n * H_ + h) * DE + t];
    __syncthreads();
    float lg[4], lmax = -1e30f;
#pragma unroll
    for (int i = 0; i < 4; ++i) {
        int m = t + i * 256;
        float a = 0.f;
#pragma unroll
        for (int d = 0; d < DE; ++d) a += qn[d] * Kt[(m * H_ + h) * DE + d];
        a *= 0.25f;                         // 1/sqrt(16)
        lg[i] = a; lmax = fmaxf(lmax, a);
    }
    red[t] = lmax; __syncthreads();
    for (int s = 128; s > 0; s >>= 1) {
        if (t < s) red[t] = fmaxf(red[t], red[t + s]);
        __syncthreads();
    }
    float gmax = red[0]; __syncthreads();
    float lsum = 0.f;
#pragma unroll
    for (int i = 0; i < 4; ++i) { lg[i] = __expf(lg[i] - gmax); lsum += lg[i]; }
    red[t] = lsum; __syncthreads();
    for (int s = 128; s > 0; s >>= 1) {
        if (t < s) red[t] += red[t + s];
        __syncthreads();
    }
    float inv = 1.f / red[0];
    float alpha = 1.f / (1.f + __expf(-*alpha_s));
#pragma unroll
    for (int i = 0; i < 4; ++i) {
        int m = t + i * 256;
        float v = alpha * Abase[(size_t)n * N_ + m] + (1.f - alpha) * lg[i] * inv;
        Ah[(size_t)n * N_ + m] = (f16)v;
    }
}

// =====================================================================
// Kernel 4: per-head node filters  Wf[n,k,l] = sum_d psi[n,d]*Fw[h,d,k,l]
// =====================================================================
__global__ void wf_kernel(const float* __restrict__ psi_emb,
                          const float* __restrict__ Fw,
                          float* __restrict__ Wf, int h) {
    int n = blockIdx.x, t = threadIdx.x;
    __shared__ float pn[DE];
    if (t < DE) pn[t] = psi_emb[n * DE + t];
    __syncthreads();
    for (int o = t; o < 4 * L_; o += 256) {
        int k = o >> 7, l = o & 127;
        float a = 0.f;
#pragma unroll
        for (int d = 0; d < DE; ++d)
            a += pn[d] * Fw[((h * DE + d) * 4 + k) * L_ + l];
        Wf[(n * 4 + k) * L_ + l] = a;
    }
}

// =====================================================================
// Kernel 5: transpose x (B,N,L) f32 -> Z0 (C,N) f16, c = b*L + l
// =====================================================================
__global__ void xtoz0_kernel(const float* __restrict__ x, f16* __restrict__ Z0) {
    __shared__ f16 tile[32][33];
    int n0 = blockIdx.x * 32, l0 = blockIdx.y * 32, b = blockIdx.z;
    int tx = threadIdx.x, ty = threadIdx.y;
#pragma unroll
    for (int i = 0; i < 32; i += 8)
        tile[ty + i][tx] = (f16)x[((size_t)b * N_ + (n0 + ty + i)) * L_ + l0 + tx];
    __syncthreads();
#pragma unroll
    for (int i = 0; i < 32; i += 8)
        Z0[(size_t)(b * L_ + l0 + ty + i) * N_ + n0 + tx] = tile[tx][ty + i];
}

// =====================================================================
// Kernel 6: WMMA GEMM  Zout[c,n] = scaleA * sum_m A[n,m]*Zin[c,m] - Sub[c,n]
// A: (N_,N_) f16 row-major;  Zin/Zout/Sub: (C_,N_) f16 c-major.
// Tile 128(n) x 128(c) x 32(m); 8 waves (2x4); each wave 4x2 WMMA tiles.
// Double-buffered LDS filled by GLOBAL_LOAD_ASYNC_TO_LDS_B128 (ASYNCcnt).
// =====================================================================
#define MT  128
#define CT  128
#define KT  32
#define LDT 40     // padded LDS row stride in halfs (80B = 20 banks)

__launch_bounds__(256)
__global__ void cheb_gemm_kernel(const f16* __restrict__ A,
                                 const f16* __restrict__ Zin,
                                 f16* __restrict__ Zout,
                                 const f16* __restrict__ Sub,
                                 float scaleA) {
    __shared__ f16 At[2][MT * LDT];
    __shared__ f16 Bt[2][CT * LDT];
    int c0 = blockIdx.x * CT;
    int n0 = blockIdx.y * MT;
    int t = threadIdx.x;
    int wave = t >> 5, lane = t & 31;
    int wm = wave >> 2;          // 0..1 -> n offset wm*64
    int wc = wave & 3;           // 0..3 -> c offset wc*32
    int lrow = lane & 15;
    int lhi  = lane >> 4;        // 0/1

    v8f acc[4][2];
    const v8f vz = {0.f, 0.f, 0.f, 0.f, 0.f, 0.f, 0.f, 0.f};
#pragma unroll
    for (int mi = 0; mi < 4; ++mi)
#pragma unroll
        for (int ni = 0; ni < 2; ++ni) acc[mi][ni] = vz;

    // staging assignment: thread t moves rows ar0 and ar0+64, 16B chunk ac0
    int ar0 = t >> 2, ac0 = t & 3;
    const unsigned SS = (unsigned)(MT * LDT * sizeof(f16));   // stage stride (bytes)
    unsigned ldsA = (unsigned)(uintptr_t)&At[0][0] + (unsigned)((ar0 * LDT + ac0 * 8) * sizeof(f16));
    unsigned ldsB = (unsigned)(uintptr_t)&Bt[0][0] + (unsigned)((ar0 * LDT + ac0 * 8) * sizeof(f16));
    const unsigned rowHalf = (unsigned)(64 * LDT * sizeof(f16)); // +64 rows in LDS
    const f16* gA = A   + (size_t)(n0 + ar0) * N_ + ac0 * 8;
    const f16* gB = Zin + (size_t)(c0 + ar0) * N_ + ac0 * 8;
    const size_t gRow64 = (size_t)64 * N_;

    // ---- prologue: issue stage 0 ----
    async_copy_b128(ldsA,           gA);
    async_copy_b128(ldsA + rowHalf, gA + gRow64);
    async_copy_b128(ldsB,           gB);
    async_copy_b128(ldsB + rowHalf, gB + gRow64);

    int stage = 0;
    for (int k0 = 0; k0 < N_; k0 += KT) {
        int s = stage & 1;
        if (k0 + KT < N_) {
            // issue stage k+1 into the other buffer (freed by last iter's barrier)
            unsigned so = (unsigned)((s ^ 1)) * SS;
            const f16* a = gA + (k0 + KT);
            const f16* b = gB + (k0 + KT);
            async_copy_b128(ldsA + so,           a);
            async_copy_b128(ldsA + so + rowHalf, a + gRow64);
            async_copy_b128(ldsB + so,           b);
            async_copy_b128(ldsB + so + rowHalf, b + gRow64);
            // wait for stage k (4 of ours still outstanding for stage k+1)
            asm volatile("s_wait_asynccnt 0x4" ::: "memory");
        } else {
            asm volatile("s_wait_asynccnt 0x0" ::: "memory");
        }
        __syncthreads();   // make all waves' async writes to buffer s visible

        const f16* AtS = &At[s][0];
        const f16* BtS = &Bt[s][0];

        // A fragments: lane<16 holds K {0..7,16..23}; lane>=16 holds {8..15,24..31}
        v16h af[4];
#pragma unroll
        for (int mi = 0; mi < 4; ++mi) {
            const f16* p = &AtS[(wm * 64 + mi * 16 + lrow) * LDT + lhi * 8];
            half8 lo = *(const half8*)p;
            half8 hi = *(const half8*)(p + 16);
            af[mi] = __builtin_shufflevector(lo, hi, 0, 1, 2, 3, 4, 5, 6, 7,
                                             8, 9, 10, 11, 12, 13, 14, 15);
        }
        // B fragments: lane<16 (col=lane) holds K {0..7,16..23}; lane>=16 holds {8..15,24..31}
        v16h bfr[2];
#pragma unroll
        for (int ni = 0; ni < 2; ++ni) {
            const f16* p = &BtS[(wc * 32 + ni * 16 + lrow) * LDT + lhi * 8];
            half8 lo = *(const half8*)p;
            half8 hi = *(const half8*)(p + 16);
            bfr[ni] = __builtin_shufflevector(lo, hi, 0, 1, 2, 3, 4, 5, 6, 7,
                                              8, 9, 10, 11, 12, 13, 14, 15);
        }
#pragma unroll
        for (int mi = 0; mi < 4; ++mi)
#pragma unroll
            for (int ni = 0; ni < 2; ++ni)
                acc[mi][ni] = __builtin_amdgcn_wmma_f32_16x16x32_f16(
                    false, af[mi], false, bfr[ni], (short)0, acc[mi][ni],
                    false, false);

        __syncthreads();   // everyone done reading buffer s before it is refilled
        ++stage;
    }

    // epilogue: lane holds column c (N of D) and 8 consecutive n rows (M of D)
#pragma unroll
    for (int mi = 0; mi < 4; ++mi) {
#pragma unroll
        for (int ni = 0; ni < 2; ++ni) {
            int c  = c0 + wc * 32 + ni * 16 + lrow;
            int nb = n0 + wm * 64 + mi * 16 + lhi * 8;
            size_t off = (size_t)c * N_ + nb;
            half8 out;
            if (Sub) {
                half8 sv = *(const half8*)&Sub[off];
#pragma unroll
                for (int i = 0; i < 8; ++i)
                    out[i] = (f16)(scaleA * acc[mi][ni][i] - (float)sv[i]);
            } else {
#pragma unroll
                for (int i = 0; i < 8; ++i)
                    out[i] = (f16)(scaleA * acc[mi][ni][i]);
            }
            *(half8*)&Zout[off] = out;
        }
    }
}

// =====================================================================
// Kernel 7: per-head contraction
// outH[h,b,n] = sum_{k,l} Zk[(b*L+l), n] * Wf[n,k,l] + bf[h,n]
// =====================================================================
__global__ void contract_kernel(const f16* __restrict__ Z0,
                                const f16* __restrict__ Z1,
                                const f16* __restrict__ Z2,
                                const f16* __restrict__ Z3,
                                const float* __restrict__ Wf,
                                const float* __restrict__ bf,
                                float* __restrict__ outH, int h) {
    int n0 = blockIdx.x * 64, b = blockIdx.y;
    int t = threadIdx.x;
    int ni = t & 63, lg = t >> 6;
    int n = n0 + ni;
    const f16* Zs[4] = {Z0, Z1, Z2, Z3};
    float a = 0.f;
#pragma unroll
    for (int k = 0; k < 4; ++k) {
        const f16* Z = Zs[k];
        const float* W = &Wf[(n * 4 + k) * L_];
        for (int l = lg; l < L_; l += 4)
            a += (float)Z[(size_t)(b * L_ + l) * N_ + n] * W[l];
    }
    __shared__ float red[4][64];
    red[lg][ni] = a;
    __syncthreads();
    if (lg == 0) {
        float s = red[0][ni] + red[1][ni] + red[2][ni] + red[3][ni] + bf[h * N_ + n];
        outH[((size_t)h * B_ + b) * N_ + n] = s;
    }
}

// =====================================================================
// Kernel 8: head mix  out[b,n] = sum_h softmax(head_mix)[h] * outH[h,b,n]
// =====================================================================
__global__ void mix_kernel(const float* __restrict__ head_mix,
                           const float* __restrict__ outH,
                           float* __restrict__ out) {
    int idx = blockIdx.x * 256 + threadIdx.x;
    float m0 = head_mix[0], m1 = head_mix[1], m2 = head_mix[2], m3 = head_mix[3];
    float mx = fmaxf(fmaxf(m0, m1), fmaxf(m2, m3));
    float e0 = __expf(m0 - mx), e1 = __expf(m1 - mx),
          e2 = __expf(m2 - mx), e3 = __expf(m3 - mx);
    float inv = 1.f / (e0 + e1 + e2 + e3);
    int b = idx / N_, n = idx % N_;
    size_t base = (size_t)b * N_ + n;
    float a = e0 * outH[base] + e1 * outH[(size_t)B_ * N_ + base] +
              e2 * outH[(size_t)2 * B_ * N_ + base] +
              e3 * outH[(size_t)3 * B_ * N_ + base];
    out[idx] = a * inv;
}

// =====================================================================
extern "C" void kernel_launch(void* const* d_in, const int* in_sizes, int n_in,
                              void* d_out, int out_size, void* d_ws, size_t ws_size,
                              hipStream_t stream) {
    const float* x        = (const float*)d_in[0];
    const float* psi_emb  = (const float*)d_in[1];
    const float* psi      = (const float*)d_in[2];
    const float* W_q      = (const float*)d_in[3];
    const float* W_k      = (const float*)d_in[4];
    const float* alpha    = (const float*)d_in[5];
    const float* F_w      = (const float*)d_in[6];
    const float* f_b      = (const float*)d_in[7];
    const float* head_mix = (const float*)d_in[8];
    float* out = (float*)d_out;

    char* ws = (char*)d_ws;
    const size_t MB = 1024 * 1024;
    f16*   Z0    = (f16*)(ws + 0 * MB);      // (C,N) f16   8 MB
    f16*   Z1    = (f16*)(ws + 8 * MB);      //             8 MB
    f16*   Z2    = (f16*)(ws + 16 * MB);     //             8 MB
    f16*   Z3    = (f16*)(ws + 24 * MB);     //             8 MB
    float* Abase = (float*)(ws + 32 * MB);   // (N,N) f32   4 MB
    f16*   Ah    = (f16*)(ws + 36 * MB);     // (N,N) f16   2 MB
    float* Qb    = (float*)(ws + 38 * MB);   // (N,H,16)    256 KB
    float* Ktb   = (float*)(ws + 38 * MB + 256 * 1024);
    float* Wf    = (float*)(ws + 39 * MB);   // (N,4,128)   2 MB
    float* bf    = (float*)(ws + 41 * MB);   // (H,N)       16 KB
    float* outH  = (float*)(ws + 41 * MB + 64 * 1024);  // (H,B,N) 512 KB

    prep_qk_kernel<<<N_, 128, 0, stream>>>(psi_emb, W_q, W_k, f_b, Qb, Ktb, bf);
    abase_kernel<<<N_, 256, 0, stream>>>(psi_emb, psi, Abase);
    xtoz0_kernel<<<dim3(N_ / 32, L_ / 32, B_), dim3(32, 8), 0, stream>>>(x, Z0);

    dim3 ggrid(C_ / CT, N_ / MT);
    for (int h = 0; h < H_; ++h) {
        ah_kernel<<<N_, 256, 0, stream>>>(Qb, Ktb, Abase, alpha, Ah, h);
        wf_kernel<<<N_, 256, 0, stream>>>(psi_emb, F_w, Wf, h);
        // Z1 = A Z0 ; Z2 = 2 A Z1 - Z0 ; Z3 = 2 A Z2 - Z1
        cheb_gemm_kernel<<<ggrid, 256, 0, stream>>>(Ah, Z0, Z1, nullptr, 1.f);
        cheb_gemm_kernel<<<ggrid, 256, 0, stream>>>(Ah, Z1, Z2, Z0, 2.f);
        cheb_gemm_kernel<<<ggrid, 256, 0, stream>>>(Ah, Z2, Z3, Z1, 2.f);
        contract_kernel<<<dim3(N_ / 64, B_), 256, 0, stream>>>(Z0, Z1, Z2, Z3,
                                                               Wf, bf, outH, h);
    }
    mix_kernel<<<(B_ * N_) / 256, 256, 0, stream>>>(head_mix, outH, out);
}
